// SlotAttention_35682588295465
// MI455X (gfx1250) — matrix operand order repaired and checked by reading, not compiled, not run on hardware
//
#include <hip/hip_runtime.h>
#include <hip/hip_bf16.h>

#define BB 8
#define NN 2048
#define DD 256
#define SS 8
#define HH 256
#define N_ITERS 3
#define EPSA 1e-8f
#define LN_EPS 1e-5f
#define UCH 16            // n-chunks for the updates reduction
#define Q_SCALE 0.0625f   // D^-0.5 = 1/16
#define MTILE 4           // M-tiles (16 rows each) per wave in the k/v GEMM

typedef __attribute__((ext_vector_type(16))) __bf16 v16bf;
typedef __attribute__((ext_vector_type(8)))  float  v8f;
static_assert(sizeof(v16bf) == 32, "v16bf must be 32 bytes");

// ---------------------------------------------------------------------------
// 0a. LayerNorm of inputs (normalization only; affines folded into weights),
//     store bf16 [B*N, D]
// ---------------------------------------------------------------------------
__global__ __launch_bounds__(256) void k_ln_inputs(const float* __restrict__ inp,
                                                   __bf16* __restrict__ xn) {
  __shared__ float s1[256], s2[256];
  const int row = blockIdx.x, t = threadIdx.x;
  float x = inp[(size_t)row * DD + t];
  s1[t] = x; s2[t] = x * x; __syncthreads();
  for (int s = 128; s > 0; s >>= 1) {
    if (t < s) { s1[t] += s1[t + s]; s2[t] += s2[t + s]; }
    __syncthreads();
  }
  const float mu = s1[0] * (1.0f / DD);
  const float var = s2[0] * (1.0f / DD) - mu * mu;
  const float rs = rsqrtf(var + LN_EPS);
  xn[(size_t)row * DD + t] = (__bf16)((x - mu) * rs);
}

// ---------------------------------------------------------------------------
// 0b. Fold per-slot LN affines into weights & biases.
//     mat 0: Wk (g_in,b_in) -> bf16 ; 1: Wv -> bf16 ; 2: Wq (g_ns,b_ns) -> f32 ;
//     3: W1 (g_pf,b_pf) -> f32.   blockIdx = mat*2048 + s*256 + o
// ---------------------------------------------------------------------------
__global__ __launch_bounds__(256) void k_fold(
    const float* __restrict__ Wq, const float* __restrict__ bq,
    const float* __restrict__ Wk, const float* __restrict__ bk,
    const float* __restrict__ Wv, const float* __restrict__ bv,
    const float* __restrict__ W1, const float* __restrict__ b1,
    const float* __restrict__ g_in, const float* __restrict__ b_in,
    const float* __restrict__ g_ns, const float* __restrict__ b_ns,
    const float* __restrict__ g_pf, const float* __restrict__ b_pf,
    __bf16* __restrict__ Wk_bf, __bf16* __restrict__ Wv_bf,
    float* __restrict__ Wq_e, float* __restrict__ W1_e,
    float* __restrict__ bk_e, float* __restrict__ bv_e,
    float* __restrict__ bq_e, float* __restrict__ b1_e) {
  __shared__ float sm[256];
  const int t = threadIdx.x;
  const int mat = blockIdx.x >> 11;
  const int r = blockIdx.x & 2047;
  const int s = r >> 8, o = r & 255;
  const float *W, *bias, *g, *bvec;
  if (mat == 0)      { W = Wk; bias = bk; g = g_in; bvec = b_in; }
  else if (mat == 1) { W = Wv; bias = bv; g = g_in; bvec = b_in; }
  else if (mat == 2) { W = Wq; bias = bq; g = g_ns; bvec = b_ns; }
  else               { W = W1; bias = b1; g = g_pf; bvec = b_pf; }
  const size_t widx = ((size_t)s * DD + o) * DD + t;  // H == D == 256
  const float w = W[widx];
  const float weff = w * g[s * DD + t];
  if (mat == 0)      Wk_bf[widx] = (__bf16)weff;
  else if (mat == 1) Wv_bf[widx] = (__bf16)weff;
  else if (mat == 2) Wq_e[widx] = weff;
  else               W1_e[widx] = weff;
  sm[t] = w * bvec[s * DD + t]; __syncthreads();
  for (int st = 128; st > 0; st >>= 1) {
    if (t < st) sm[t] += sm[t + st];
    __syncthreads();
  }
  if (t == 0) {
    const float be = bias[s * DD + o] + sm[0];
    if (mat == 0)      bk_e[s * DD + o] = be;
    else if (mat == 1) bv_e[s * DD + o] = be;
    else if (mat == 2) bq_e[s * DD + o] = be;
    else               b1_e[s * DD + o] = be;
  }
}

// ---------------------------------------------------------------------------
// 0c. slots = mu + exp(logsigma) * noise   [B,S,D]
// ---------------------------------------------------------------------------
__global__ void k_slots_init(const float* __restrict__ mu, const float* __restrict__ ls,
                             const float* __restrict__ noise, float* __restrict__ slots) {
  const int i = blockIdx.x * blockDim.x + threadIdx.x;
  if (i < BB * SS * DD) {
    const int sd = i % (SS * DD);
    slots[i] = mu[sd] + expf(ls[sd]) * noise[i];
  }
}

// ---------------------------------------------------------------------------
// 1. k/v projection GEMM on WMMA (bf16 x bf16 -> f32 accum):
//    k[b,s,n,o] = sum_d xn[b,n,d] * Wk_eff[s,o,d] + bk_eff[s,o]
//    One wave computes a 64(M)x16(o) strip of BOTH k and v: 4 M-tiles reuse
//    each Bk/Bv fragment (4 WMMAs per B load pair) -> 1.5 b128 loads / WMMA.
//    K loop: 256 in steps of 32 -> 64 WMMAs per wave.
// ---------------------------------------------------------------------------
__global__ __launch_bounds__(256) void k_kv_gemm(
    const __bf16* __restrict__ xn,        // [B*N, D]
    const __bf16* __restrict__ Wkb,       // [S, D, D] (o-major, d contiguous)
    const __bf16* __restrict__ Wvb,
    const float* __restrict__ bk_e, const float* __restrict__ bv_e,
    __bf16* __restrict__ kout,            // [B, S, N, D]
    __bf16* __restrict__ vout) {
  const int OT = DD / 16;                    // 16 o-tiles
  const int MG = (BB * NN) / (16 * MTILE);   // 256 m-groups of 64 rows
  const int gw = blockIdx.x * 8 + (threadIdx.x >> 5);
  const int lane = threadIdx.x & 31;
  const int ot = gw % OT;
  const int tmp = gw / OT;
  const int mg = tmp % MG;
  const int s = tmp / MG;
  const int h = lane >> 4, lm = lane & 15;

  // Single A base per lane; M-tile i is an immediate offset of i*16*DD elements.
  const __bf16* arow = xn + (size_t)(mg * 64 + lm) * DD;
  const int o = ot * 16 + lm;
  const __bf16* krow = Wkb + ((size_t)s * DD + o) * DD;
  const __bf16* vrow = Wvb + ((size_t)s * DD + o) * DD;

  v8f ck[MTILE] = {};
  v8f cv[MTILE] = {};
#pragma unroll
  for (int d0 = 0; d0 < DD; d0 += 32) {
    // B 32x16 bf16: lane n+16h holds K = d0 + 16h .. d0 + 16h + 15 (contiguous)
    union { v16bf v; uint4 q[2]; } bkf, bvf;
    const __bf16* kp = krow + d0 + 16 * h;
    bkf.q[0] = *(const uint4*)(kp);
    bkf.q[1] = *(const uint4*)(kp + 8);
    const __bf16* vp = vrow + d0 + 16 * h;
    bvf.q[0] = *(const uint4*)(vp);
    bvf.q[1] = *(const uint4*)(vp + 8);
    // A 16x32 bf16: lane m+16h holds K = d0 + 16h .. d0 + 16h + 15 (contiguous)
    union { v16bf v; uint4 q[2]; } af[MTILE];
#pragma unroll
    for (int i = 0; i < MTILE; i++) {
      const __bf16* ap = arow + (size_t)i * 16 * DD + d0 + 16 * h;
      af[i].q[0] = *(const uint4*)(ap);
      af[i].q[1] = *(const uint4*)(ap + 8);
    }
#pragma unroll
    for (int i = 0; i < MTILE; i++) {
      ck[i] = __builtin_amdgcn_wmma_f32_16x16x32_bf16(false, af[i].v, false, bkf.v,
                                                      (short)0, ck[i], false, false);
      cv[i] = __builtin_amdgcn_wmma_f32_16x16x32_bf16(false, af[i].v, false, bvf.v,
                                                      (short)0, cv[i], false, false);
    }
  }
  const float bias_k = bk_e[s * DD + o];
  const float bias_v = bv_e[s * DD + o];
  // D layout: VGPR j holds row m = 8*(lane/16)+j, col n = lane%16
  const int b = (mg * 64) / NN;              // 64 | N -> group never spans batches
  const int nbase = (mg * 64) % NN;
#pragma unroll
  for (int i = 0; i < MTILE; i++) {
    const size_t base =
        (((size_t)b * SS + s) * NN + nbase + i * 16 + 8 * h) * DD + o;
#pragma unroll
    for (int j = 0; j < 8; j++) {
      kout[base + (size_t)j * DD] = (__bf16)(ck[i][j] + bias_k);
      vout[base + (size_t)j * DD] = (__bf16)(cv[i][j] + bias_v);
    }
  }
}

// ---------------------------------------------------------------------------
// 2. q = (LN(slots) @ Wq_eff^T + bq_eff) * SCALE   (one block per (b,s))
// ---------------------------------------------------------------------------
__global__ __launch_bounds__(256) void k_q(const float* __restrict__ slots,
                                           const float* __restrict__ Wq_e,
                                           const float* __restrict__ bq_e,
                                           float* __restrict__ q) {
  __shared__ float s1[256], s2[256], sn[DD];
  const int bs = blockIdx.x, t = threadIdx.x, s = bs % SS;
  const float x = slots[(size_t)bs * DD + t];
  s1[t] = x; s2[t] = x * x; __syncthreads();
  for (int st = 128; st > 0; st >>= 1) {
    if (t < st) { s1[t] += s1[t + st]; s2[t] += s2[t + st]; }
    __syncthreads();
  }
  const float mu = s1[0] * (1.0f / DD);
  const float var = s2[0] * (1.0f / DD) - mu * mu;
  const float rs = rsqrtf(var + LN_EPS);
  sn[t] = (x - mu) * rs;
  __syncthreads();
  const float* wrow = Wq_e + ((size_t)s * DD + t) * DD;
  float acc = bq_e[s * DD + t];
  for (int d = 0; d < DD; d++) acc += sn[d] * wrow[d];
  q[(size_t)bs * DD + t] = acc * Q_SCALE;
}

// ---------------------------------------------------------------------------
// 3. dots[b,s,n] = q[b,s,:] . k[b,s,n,:]  (wave per n, lane covers 8 d's)
// ---------------------------------------------------------------------------
__global__ __launch_bounds__(256) void k_dots(const float* __restrict__ q,
                                              const __bf16* __restrict__ kk,
                                              float* __restrict__ dots) {
  const int bs = blockIdx.x / (NN / 64);
  const int nb = blockIdx.x % (NN / 64);
  const int wave = threadIdx.x >> 5, lane = threadIdx.x & 31;
  const float* qrow = q + (size_t)bs * DD + lane * 8;
  float qv[8];
#pragma unroll
  for (int j = 0; j < 8; j++) qv[j] = qrow[j];
#pragma unroll
  for (int i = 0; i < 8; i++) {
    const int n = nb * 64 + wave * 8 + i;
    const __bf16* kr = kk + ((size_t)bs * NN + n) * DD + lane * 8;
    float p = 0.f;
#pragma unroll
    for (int j = 0; j < 8; j++) p += (float)kr[j] * qv[j];
#pragma unroll
    for (int off = 16; off > 0; off >>= 1) p += __shfl_xor(p, off);
    if (lane == 0) dots[(size_t)bs * NN + n] = p;
  }
}

// ---------------------------------------------------------------------------
// 4. softmax over slot axis (S=8) + EPS  -> attn_ori
// ---------------------------------------------------------------------------
__global__ __launch_bounds__(256) void k_softmax(const float* __restrict__ dots,
                                                 float* __restrict__ attn) {
  const int i = blockIdx.x * 256 + threadIdx.x;  // over B*N
  const int b = i / NN, n = i % NN;
  float x[SS], m = -1e30f;
#pragma unroll
  for (int s = 0; s < SS; s++) {
    x[s] = dots[((size_t)b * SS + s) * NN + n];
    m = fmaxf(m, x[s]);
  }
  float sum = 0.f;
#pragma unroll
  for (int s = 0; s < SS; s++) { x[s] = expf(x[s] - m); sum += x[s]; }
  const float inv = 1.0f / sum;
#pragma unroll
  for (int s = 0; s < SS; s++)
    attn[((size_t)b * SS + s) * NN + n] = x[s] * inv + EPSA;
}

// ---------------------------------------------------------------------------
// 5. denom[b,s] = sum_n attn_ori[b,s,n]
// ---------------------------------------------------------------------------
__global__ __launch_bounds__(256) void k_rowsum(const float* __restrict__ attn,
                                                float* __restrict__ denom) {
  __shared__ float sm[256];
  const int bs = blockIdx.x, t = threadIdx.x;
  float acc = 0.f;
  for (int n = t; n < NN; n += 256) acc += attn[(size_t)bs * NN + n];
  sm[t] = acc; __syncthreads();
  for (int st = 128; st > 0; st >>= 1) {
    if (t < st) sm[t] += sm[t + st];
    __syncthreads();
  }
  if (t == 0) denom[bs] = sm[0];
}

// ---------------------------------------------------------------------------
// 6. updates: partial sums over n-chunks, then reduce (+ 1/denom)
// ---------------------------------------------------------------------------
__global__ __launch_bounds__(256) void k_upart(const __bf16* __restrict__ v,
                                               const float* __restrict__ attn,
                                               float* __restrict__ part) {
  const int bs = blockIdx.x / UCH;
  const int c = blockIdx.x % UCH;
  const int d = threadIdx.x;
  const int n0 = c * (NN / UCH);
  float acc = 0.f;
  for (int j = 0; j < NN / UCH; j++) {
    const int n = n0 + j;
    const float a = attn[(size_t)bs * NN + n];
    acc += (float)v[((size_t)bs * NN + n) * DD + d] * a;
  }
  part[((size_t)bs * UCH + c) * DD + d] = acc;
}

__global__ void k_ureduce(const float* __restrict__ part,
                          const float* __restrict__ denom,
                          float* __restrict__ upd) {
  const int i = blockIdx.x * blockDim.x + threadIdx.x;  // B*S*D
  const int bs = i / DD, d = i % DD;
  float acc = 0.f;
#pragma unroll
  for (int c = 0; c < UCH; c++) acc += part[((size_t)bs * UCH + c) * DD + d];
  upd[i] = acc / denom[bs];
}

// ---------------------------------------------------------------------------
// 7. per-slot GRUCell (block per (b,s), 768 threads: one per gate output)
// ---------------------------------------------------------------------------
__global__ __launch_bounds__(768) void k_gru(const float* __restrict__ upd,
                                             const float* __restrict__ W_ih,
                                             const float* __restrict__ b_ih,
                                             const float* __restrict__ W_hh,
                                             const float* __restrict__ b_hh,
                                             float* __restrict__ slots) {
  __shared__ float u[DD], hprev[DD], gi[3 * DD], gh[3 * DD];
  const int bs = blockIdx.x, t = threadIdx.x, s = bs % SS;
  if (t < DD) {
    u[t] = upd[(size_t)bs * DD + t];
    hprev[t] = slots[(size_t)bs * DD + t];
  }
  __syncthreads();
  {
    const float* wi = W_ih + ((size_t)s * 3 * DD + t) * DD;
    const float* wh = W_hh + ((size_t)s * 3 * DD + t) * DD;
    float a = b_ih[s * 3 * DD + t];
    float c = b_hh[s * 3 * DD + t];
    for (int d = 0; d < DD; d++) { a += u[d] * wi[d]; c += hprev[d] * wh[d]; }
    gi[t] = a; gh[t] = c;
  }
  __syncthreads();
  if (t < DD) {
    const float r = 1.0f / (1.0f + expf(-(gi[t] + gh[t])));
    const float z = 1.0f / (1.0f + expf(-(gi[DD + t] + gh[DD + t])));
    const float nn = tanhf(gi[2 * DD + t] + r * gh[2 * DD + t]);
    slots[(size_t)bs * DD + t] = (1.0f - z) * nn + z * hprev[t];
  }
}

// ---------------------------------------------------------------------------
// 8. per-slot MLP residual: slots += W2 @ relu(W1_eff @ LN(slots) + b1_eff) + b2
// ---------------------------------------------------------------------------
__global__ __launch_bounds__(256) void k_mlp(const float* __restrict__ W1_e,
                                             const float* __restrict__ b1_e,
                                             const float* __restrict__ W2,
                                             const float* __restrict__ b2,
                                             float* __restrict__ slots) {
  __shared__ float s1[256], s2[256], sn[DD], h1[HH], sx[DD];
  const int bs = blockIdx.x, t = threadIdx.x, s = bs % SS;
  const float x = slots[(size_t)bs * DD + t];
  sx[t] = x; s1[t] = x; s2[t] = x * x; __syncthreads();
  for (int st = 128; st > 0; st >>= 1) {
    if (t < st) { s1[t] += s1[t + st]; s2[t] += s2[t + st]; }
    __syncthreads();
  }
  const float mu = s1[0] * (1.0f / DD);
  const float var = s2[0] * (1.0f / DD) - mu * mu;
  const float rs = rsqrtf(var + LN_EPS);
  sn[t] = (x - mu) * rs;
  __syncthreads();
  const float* w1 = W1_e + ((size_t)s * HH + t) * DD;
  float a = b1_e[s * HH + t];
  for (int d = 0; d < DD; d++) a += sn[d] * w1[d];
  h1[t] = fmaxf(a, 0.0f);
  __syncthreads();
  const float* w2 = W2 + ((size_t)s * DD + t) * HH;
  float o = b2[s * DD + t];
  for (int j = 0; j < HH; j++) o += h1[j] * w2[j];
  slots[(size_t)bs * DD + t] = sx[t] + o;
}

// ---------------------------------------------------------------------------
// 9. pack outputs: [slots (B*S*D) | attn_ori (B*S*N)]
// ---------------------------------------------------------------------------
__global__ void k_out(const float* __restrict__ slots, const float* __restrict__ attn,
                      float* __restrict__ out) {
  const int i = blockIdx.x * blockDim.x + threadIdx.x;
  const int NS = BB * SS * DD;
  const int NA = BB * SS * NN;
  if (i < NS) out[i] = slots[i];
  else if (i < NS + NA) out[i] = attn[i - NS];
}

// ---------------------------------------------------------------------------
extern "C" void kernel_launch(void* const* d_in, const int* in_sizes, int n_in,
                              void* d_out, int out_size, void* d_ws, size_t ws_size,
                              hipStream_t stream) {
  (void)in_sizes; (void)n_in; (void)out_size; (void)ws_size;
  const float* inputs   = (const float*)d_in[0];
  const float* noise    = (const float*)d_in[1];
  const float* mu       = (const float*)d_in[2];
  const float* ls       = (const float*)d_in[3];
  const float* g_in     = (const float*)d_in[4];
  const float* b_in     = (const float*)d_in[5];
  const float* g_ns     = (const float*)d_in[6];
  const float* b_ns     = (const float*)d_in[7];
  const float* g_pf     = (const float*)d_in[8];
  const float* b_pf     = (const float*)d_in[9];
  const float* Wq       = (const float*)d_in[10];
  const float* bq       = (const float*)d_in[11];
  const float* Wk       = (const float*)d_in[12];
  const float* bk       = (const float*)d_in[13];
  const float* Wv       = (const float*)d_in[14];
  const float* bv       = (const float*)d_in[15];
  const float* W_ih     = (const float*)d_in[16];
  const float* b_ih     = (const float*)d_in[17];
  const float* W_hh     = (const float*)d_in[18];
  const float* b_hh     = (const float*)d_in[19];
  const float* W1       = (const float*)d_in[20];
  const float* b1       = (const float*)d_in[21];
  const float* W2       = (const float*)d_in[22];
  const float* b2       = (const float*)d_in[23];
  float* out = (float*)d_out;

  // --- workspace bump allocator ---
  char* ws = (char*)d_ws;
  size_t off = 0;
  auto alloc = [&](size_t bytes) -> void* {
    void* p = ws + off;
    off = (off + bytes + 255) & ~(size_t)255;
    return p;
  };
  __bf16* xn    = (__bf16*)alloc((size_t)BB * NN * DD * 2);
  __bf16* Wk_bf = (__bf16*)alloc((size_t)SS * DD * DD * 2);
  __bf16* Wv_bf = (__bf16*)alloc((size_t)SS * DD * DD * 2);
  float*  Wq_e  = (float*)alloc((size_t)SS * DD * DD * 4);
  float*  W1_e  = (float*)alloc((size_t)SS * HH * DD * 4);
  float*  bk_e  = (float*)alloc((size_t)SS * DD * 4);
  float*  bv_e  = (float*)alloc((size_t)SS * DD * 4);
  float*  bq_e  = (float*)alloc((size_t)SS * DD * 4);
  float*  b1_e  = (float*)alloc((size_t)SS * HH * 4);
  __bf16* kbuf  = (__bf16*)alloc((size_t)BB * SS * NN * DD * 2);
  __bf16* vbuf  = (__bf16*)alloc((size_t)BB * SS * NN * DD * 2);
  float*  slots = (float*)alloc((size_t)BB * SS * DD * 4);
  float*  qb    = (float*)alloc((size_t)BB * SS * DD * 4);
  float*  dotsb = (float*)alloc((size_t)BB * SS * NN * 4);
  float*  attnb = (float*)alloc((size_t)BB * SS * NN * 4);
  float*  denom = (float*)alloc((size_t)BB * SS * 4);
  float*  upart = (float*)alloc((size_t)BB * SS * UCH * DD * 4);
  float*  updb  = (float*)alloc((size_t)BB * SS * DD * 4);

  // --- one-time prep ---
  k_ln_inputs<<<BB * NN, 256, 0, stream>>>(inputs, xn);
  k_fold<<<4 * SS * DD, 256, 0, stream>>>(Wq, bq, Wk, bk, Wv, bv, W1, b1,
                                          g_in, b_in, g_ns, b_ns, g_pf, b_pf,
                                          Wk_bf, Wv_bf, Wq_e, W1_e,
                                          bk_e, bv_e, bq_e, b1_e);
  k_slots_init<<<(BB * SS * DD + 255) / 256, 256, 0, stream>>>(mu, ls, noise, slots);
  // big WMMA GEMM: 8 slots * 256 m-groups * 16 o-tiles waves / 8 waves-per-block
  k_kv_gemm<<<(SS * ((BB * NN) / (16 * MTILE)) * (DD / 16)) / 8, 256, 0, stream>>>(
      xn, Wk_bf, Wv_bf, bk_e, bv_e, kbuf, vbuf);

  // --- iteration loop ---
  for (int it = 0; it < N_ITERS; it++) {
    k_q<<<BB * SS, 256, 0, stream>>>(slots, Wq_e, bq_e, qb);
    k_dots<<<BB * SS * (NN / 64), 256, 0, stream>>>(qb, kbuf, dotsb);
    k_softmax<<<(BB * NN) / 256, 256, 0, stream>>>(dotsb, attnb);
    k_rowsum<<<BB * SS, 256, 0, stream>>>(attnb, denom);
    k_upart<<<BB * SS * UCH, 256, 0, stream>>>(vbuf, attnb, upart);
    k_ureduce<<<(BB * SS * DD + 255) / 256, 256, 0, stream>>>(upart, denom, updb);
    k_gru<<<BB * SS, 768, 0, stream>>>(updb, W_ih, b_ih, W_hh, b_hh, slots);
    k_mlp<<<BB * SS, 256, 0, stream>>>(W1_e, b1_e, W2, b2, slots);
  }

  k_out<<<(BB * SS * DD + BB * SS * NN + 255) / 256, 256, 0, stream>>>(slots, attnb, out);
}
